// MultiGATLayer_24627342475868
// MI455X (gfx1250) — matrix operand order repaired and checked by reading, not compiled, not run on hardware
//
#include <hip/hip_runtime.h>
#include <hip/hip_bf16.h>

typedef __attribute__((ext_vector_type(16))) _Float16 v16h;
typedef __attribute__((ext_vector_type(8)))  _Float16 v8h;
typedef __attribute__((ext_vector_type(4)))  _Float16 v4h;
typedef __attribute__((ext_vector_type(8)))  float    v8f;
typedef __attribute__((ext_vector_type(4)))  float    v4f;

#define N_NODES 20000
#define DEG     16
#define F_IN    128
#define F_OUT   128
#define H_HEADS 4
#define HF      (H_HEADS * F_OUT)   // 512
#define LDS_STRIDE 136              // 128 + 8 halves pad -> conflict-free strided row reads

// ---------------------------------------------------------------------------
// Kernel 0a: convert features f32 -> f16 (row-major, same layout)
// ---------------------------------------------------------------------------
__global__ void gat_cvt_feat(const float* __restrict__ f, _Float16* __restrict__ o) {
    int t = blockIdx.x * blockDim.x + threadIdx.x;       // one thread = 4 elems
    int i = t * 4;
    if (i >= N_NODES * F_IN) return;
    v4f  v = *(const v4f*)(f + i);
    v4h  h;
    h.x = (_Float16)v.x; h.y = (_Float16)v.y; h.z = (_Float16)v.z; h.w = (_Float16)v.w;
    *(v4h*)(o + i) = h;
}

// ---------------------------------------------------------------------------
// Kernel 0b: convert W f32 -> f16, transposed to Wt[h][out][in] so WMMA B
// fragments are contiguous along K.
// ---------------------------------------------------------------------------
__global__ void gat_cvt_w(const float* __restrict__ w, _Float16* __restrict__ wt) {
    int t = blockIdx.x * blockDim.x + threadIdx.x;       // t = h*16384 + i*128 + o
    if (t >= H_HEADS * F_IN * F_OUT) return;
    int h  = t >> 14;
    int rm = t & 16383;
    int i  = rm >> 7;
    int o  = rm & 127;
    wt[((h * F_OUT) + o) * F_IN + i] = (_Float16)w[t];
}

// ---------------------------------------------------------------------------
// Kernel 1: h[n][head][f] = features x W + bias   (WMMA f16 -> f32 accum)
// One block = 16-node strip, 8 waves; wave w -> head = w>>1, col half = w&1
// (64 cols = 4 tiles of 16). K loop: 4 x v_wmma_f32_16x16x32_f16 per tile.
// ---------------------------------------------------------------------------
__global__ void __launch_bounds__(256)
gat_gemm_wmma(const _Float16* __restrict__ Ff16,
              const _Float16* __restrict__ Wt,
              const float*    __restrict__ bW,
              float*          __restrict__ hbuf) {
    __shared__ __align__(16) _Float16 sA[16 * LDS_STRIDE];

    const int tid  = threadIdx.x;
    const int n0   = blockIdx.x * 16;

    // cooperative stage of the 16x128 f16 feature strip into LDS (16B/thread)
    {
        int row = tid >> 4;          // 0..15
        int seg = tid & 15;          // 0..15, 8 halves each
        const v8h* src = (const v8h*)(Ff16 + (n0 + row) * F_IN + seg * 8);
        *(v8h*)(sA + row * LDS_STRIDE + seg * 8) = *src;
    }
    __syncthreads();

    const int wave = tid >> 5;
    const int lane = tid & 31;
    const int head = wave >> 1;
    const int colsBase = (wave & 1) * 64;
    const int hi = lane >> 4;        // half-wave select
    const int ln = lane & 15;

    v8f c[4] = {};

    #pragma unroll
    for (int k0 = 0; k0 < F_IN; k0 += 32) {
        // ---- A fragment (16x32 f16): K(j) = 16*(j>=8) + hi*8 + (j&7)
        const _Float16* ap = sA + ln * LDS_STRIDE + k0 + hi * 8;
        v8h alo = *(const v8h*)(ap);
        v8h ahi = *(const v8h*)(ap + 16);
        v16h a;
        #pragma unroll
        for (int i = 0; i < 8; ++i) { a[i] = alo[i]; a[i + 8] = ahi[i]; }

        #pragma unroll
        for (int t = 0; t < 4; ++t) {
            // ---- B fragment (32x16 f16): K(j) = 16*(j>=8) + hi*8 + (j&7), N = ln
            int col = colsBase + t * 16 + ln;
            const _Float16* wp = Wt + ((head * F_OUT) + col) * F_IN + k0 + hi * 8;
            v8h b0 = *(const v8h*)(wp);
            v8h b1 = *(const v8h*)(wp + 16);
            v16h b;
            #pragma unroll
            for (int i = 0; i < 8; ++i) { b[i] = b0[i]; b[i + 8] = b1[i]; }

            c[t] = __builtin_amdgcn_wmma_f32_16x16x32_f16(
                false, a, false, b, (short)0, c[t], false, false);
        }
    }

    // bias + writeback: C VGPR r -> node = n0 + r + 8*hi, col = colsBase+t*16+ln
    #pragma unroll
    for (int t = 0; t < 4; ++t) {
        int col = colsBase + t * 16 + ln;
        float bias = bW[head * F_OUT + col];
        #pragma unroll
        for (int r = 0; r < 8; ++r) {
            int node = n0 + r + 8 * hi;
            hbuf[node * HF + head * F_OUT + col] = c[t][r] + bias;
        }
    }
}

// ---------------------------------------------------------------------------
// Kernel 2: score_self/score_nb per (node, head). One wave per node.
// ---------------------------------------------------------------------------
__global__ void __launch_bounds__(256)
gat_scores(const float* __restrict__ hbuf,
           const float* __restrict__ a_w,
           float* __restrict__ ss, float* __restrict__ sn) {
    const int wave = threadIdx.x >> 5;
    const int lane = threadIdx.x & 31;
    const int n = blockIdx.x * 8 + wave;
    if (n >= N_NODES) return;

    #pragma unroll
    for (int h = 0; h < H_HEADS; ++h) {
        const float* hp = hbuf + n * HF + h * F_OUT;
        float ps = 0.f, pn = 0.f;
        #pragma unroll
        for (int i = 0; i < 4; ++i) {
            int f = i * 32 + lane;
            float v = hp[f];
            ps += v * a_w[h * 2 * F_OUT + f];
            pn += v * a_w[h * 2 * F_OUT + F_OUT + f];
        }
        #pragma unroll
        for (int off = 16; off > 0; off >>= 1) {
            ps += __shfl_xor(ps, off, 32);
            pn += __shfl_xor(pn, off, 32);
        }
        if (lane == 0) {
            ss[n * H_HEADS + h] = ps;
            sn[n * H_HEADS + h] = pn;
        }
    }
}

// ---------------------------------------------------------------------------
// Kernel 3: softmax over {16 neighbors + self} + weighted aggregation.
// One wave per (node, head). Lanes 0..16 own logits; all 32 lanes own 4
// contiguous output features each (float4 gather of neighbor h rows).
// ---------------------------------------------------------------------------
__global__ void __launch_bounds__(256)
gat_attn_agg(const int*   __restrict__ adj,
             const float* __restrict__ hbuf,
             const float* __restrict__ ss,
             const float* __restrict__ sn,
             const float* __restrict__ a_b,
             float* __restrict__ out) {
    const int wave = threadIdx.x >> 5;
    const int lane = threadIdx.x & 31;
    const int wid  = blockIdx.x * 8 + wave;
    if (wid >= N_NODES * H_HEADS) return;
    const int n = wid >> 2;
    const int h = wid & 3;

    // --- logits for the 17 edges on lanes 0..16
    int nbr = n;
    float lg = -1e30f;
    if (lane < DEG + 1) {
        nbr = (lane < DEG) ? adj[n * DEG + lane] : n;
        float z = ss[n * H_HEADS + h] + sn[nbr * H_HEADS + h] + a_b[h];
        lg = (z > 0.f) ? z : 0.2f * z;             // leaky relu, slope 0.2
    }

    // --- softmax over 17 active lanes (wave reductions)
    float m = lg;
    #pragma unroll
    for (int off = 16; off > 0; off >>= 1) {
        float o = __shfl_xor(m, off, 32);
        m = (o > m) ? o : m;
    }
    float e = (lane < DEG + 1) ? __expf(lg - m) : 0.f;
    float s = e;
    #pragma unroll
    for (int off = 16; off > 0; off >>= 1) s += __shfl_xor(s, off, 32);
    float attn = e / s;

    // --- aggregate: out[n,h,f] = sum_d attn_d * h[nbr_d, h, f]
    v4f acc = {};
    #pragma unroll
    for (int d = 0; d < DEG + 1; ++d) {
        float a_d = __shfl(attn, d, 32);
        int   n_d = __shfl(nbr,  d, 32);
        v4f v = *(const v4f*)(hbuf + n_d * HF + h * F_OUT + lane * 4);
        acc += a_d * v;
    }
    *(v4f*)(out + n * HF + h * F_OUT + lane * 4) = acc;
}

// ---------------------------------------------------------------------------
extern "C" void kernel_launch(void* const* d_in, const int* in_sizes, int n_in,
                              void* d_out, int out_size, void* d_ws, size_t ws_size,
                              hipStream_t stream) {
    const int*   adj  = (const int*)  d_in[0];
    const float* feat = (const float*)d_in[1];
    const float* W    = (const float*)d_in[2];
    const float* bW   = (const float*)d_in[3];
    const float* a_w  = (const float*)d_in[4];
    const float* a_b  = (const float*)d_in[5];
    float* out = (float*)d_out;

    // workspace carve (256B aligned slices)
    char* ws = (char*)d_ws;
    size_t off = 0;
    _Float16* Ff16 = (_Float16*)(ws + off); off += (size_t)N_NODES * F_IN * 2;       // 5,120,000
    _Float16* Wt   = (_Float16*)(ws + off); off += (size_t)H_HEADS * F_IN * F_OUT*2; //   131,072
    float*    hbuf = (float*)   (ws + off); off += (size_t)N_NODES * HF * 4;         // 40,960,000
    float*    ssb  = (float*)   (ws + off); off += (size_t)N_NODES * H_HEADS * 4;    //   320,000
    float*    snb  = (float*)   (ws + off); off += (size_t)N_NODES * H_HEADS * 4;    //   320,000
    (void)ws_size; (void)in_sizes; (void)n_in; (void)out_size;

    // 0a: features -> f16
    gat_cvt_feat<<<(N_NODES * F_IN / 4 + 255) / 256, 256, 0, stream>>>(feat, Ff16);
    // 0b: W -> f16 transposed
    gat_cvt_w<<<(H_HEADS * F_IN * F_OUT + 255) / 256, 256, 0, stream>>>(W, Wt);
    // 1: WMMA GEMM + bias
    gat_gemm_wmma<<<N_NODES / 16, 256, 0, stream>>>(Ff16, Wt, bW, hbuf);
    // 2: attention scores
    gat_scores<<<(N_NODES + 7) / 8, 256, 0, stream>>>(hbuf, a_w, ssb, snb);
    // 3: softmax + aggregation
    gat_attn_agg<<<(N_NODES * H_HEADS + 7) / 8, 256, 0, stream>>>(adj, hbuf, ssb, snb, a_b, out);
}
